// Encoder_61435212202675
// MI455X (gfx1250) — matrix operand order repaired and checked by reading, not compiled, not run on hardware
//
#include <hip/hip_runtime.h>
#include <math.h>

typedef __attribute__((ext_vector_type(16))) _Float16 v16h;
typedef __attribute__((ext_vector_type(8)))  float    v8f;
typedef __attribute__((ext_vector_type(2)))  _Float16 h2;
typedef __attribute__((ext_vector_type(2)))  float    f2;

// ---------------------------------------------------------------------------
// Conv3d implicit GEMM, f16 WMMA, f32 accumulate.
//   M = Cout, N = B*OD*OH*OW, K = Cin*KC^3 zero-padded to KKpad (mult of 32)
//   in the pre-packed f16 weight buffer. One wave computes a (MT*16)x16 C
//   tile: the im2col B fragment is gathered once per K-step and shared by MT
//   WMMAs. Grid is exact (Cout == gridDim.y * MT * 16), so the K-loop body is
//   branch-free straight-line code. All offsets fit in int32.
// ---------------------------------------------------------------------------
template<int KC, int MT>
__global__ void conv3d_wmma(const _Float16* __restrict__ X, const _Float16* __restrict__ Wt,
                            const float* __restrict__ bias, float* __restrict__ Y,
                            int Bn, int Cin, int ID, int IH, int IW,
                            int Cout, int OD, int OH, int OW,
                            int sd, int sh, int sw, int pd, int KKpad)
{
    constexpr int KC2 = KC * KC, KC3 = KC2 * KC;
    const int lane = threadIdx.x;          // wave32
    const int hi = lane >> 4, lo = lane & 15;
    const int Ntot = Bn * OD * OH * OW;
    const int nl = blockIdx.x * 16 + lo;
    const int mbase = blockIdx.y * (MT * 16);
    const int nok = (nl < Ntot) ? 1 : 0;

    int b, od, oh, ow;
    {
        int t = nok ? nl : 0;
        ow = t % OW; t /= OW;
        oh = t % OH; t /= OH;
        od = t % OD; b = t / OD;
    }
    const int id0 = od * sd - pd, ih0 = oh * sh - pd, iw0 = ow * sw - pd;
    const int planeIn = IH * IW;
    const int chanIn  = ID * planeIn;
    const int bbase   = b * Cin * chanIn;
    const h2* W2 = (const h2*)Wt;

    v8f acc[MT] = {};

    for (int k0 = 0; k0 < KKpad; k0 += 32) {
        {   // stream next weight chunk (global_prefetch_b8)
            int kpf = (k0 + 32 < KKpad) ? (k0 + 32) : 0;
            __builtin_prefetch(Wt + (mbase + lo) * KKpad + kpf, 0, 3);
        }
        // ---- B fragment (im2col gather), branchless clamp + select ----
        v16h bf;
        #pragma unroll
        for (int v = 0; v < 8; ++v) {
            #pragma unroll
            for (int hh = 0; hh < 2; ++hh) {
                int kk = k0 + hi * 8 + (v >> 2) * 16 + (v & 3) * 2 + hh;
                int ci = kk / KC3;  int r = kk - ci * KC3;
                int kd = r / KC2;   r -= kd * KC2;
                int kh = r / KC;    int kw = r - kh * KC;
                int id = id0 + kd, ih = ih0 + kh, iw = iw0 + kw;
                int ok = nok & (ci < Cin ? 1 : 0)
                             & ((unsigned)id < (unsigned)ID ? 1 : 0)
                             & ((unsigned)ih < (unsigned)IH ? 1 : 0)
                             & ((unsigned)iw < (unsigned)IW ? 1 : 0);
                int idx = bbase + ci * chanIn + id * planeIn + ih * IW + iw;
                idx = ok ? idx : 0;                 // in-bounds dummy address
                _Float16 xv = X[idx];               // unconditional load
                bf[2 * v + hh] = ok ? xv : (_Float16)0.f;
            }
        }
        // ---- MT M-subtiles share the B fragment ----
        #pragma unroll
        for (int mt = 0; mt < MT; ++mt) {
            // row pointer + compile-time element index -> immediate offsets
            const h2* ar = W2 + (((mbase + mt * 16 + lo) * KKpad + k0 + hi * 8) >> 1);
            v16h af;
            #pragma unroll
            for (int v = 0; v < 8; ++v) {
                h2 p = ar[(v >> 2) * 8 + (v & 3)];
                af[2 * v] = p[0]; af[2 * v + 1] = p[1];
            }
            acc[mt] = __builtin_amdgcn_wmma_f32_16x16x32_f16(
                false, af, false, bf, (short)0, acc[mt], false, false);
        }
    }

    if (nok) {
        const int obase = b * Cout * (OD * OH * OW) + (od * OH + oh) * OW + ow;
        const int oplane = OD * OH * OW;
        #pragma unroll
        for (int mt = 0; mt < MT; ++mt) {
            #pragma unroll
            for (int r = 0; r < 8; ++r) {
                int m = mbase + mt * 16 + r + hi * 8;   // always < Cout
                Y[obase + m * oplane] = acc[mt][r] + bias[m];
            }
        }
    }
}

// Pack conv weights [Cout][KK] f32 -> [Cout][KKpad] f16 (zero-padded K)
__global__ void pack_w_f16(const float* __restrict__ W, _Float16* __restrict__ Wp,
                           int KK, int KKpad, int total)
{
    int i = blockIdx.x * blockDim.x + threadIdx.x;
    if (i >= total) return;
    int co = i / KKpad; int kk = i - co * KKpad;
    Wp[i] = (kk < KK) ? (_Float16)W[co * KK + kk] : (_Float16)0.f;
}

// ---------------------------------------------------------------------------
// BatchNorm training stats + apply
// ---------------------------------------------------------------------------
__global__ void bn_stats(const float* __restrict__ Y, float* __restrict__ mean,
                         float* __restrict__ invstd, int Bn, int Cout, int plane)
{
    const int c = blockIdx.x;
    const int tid = threadIdx.x;
    const int total = Bn * plane;
    float s = 0.f, ss = 0.f;
    for (int i = tid; i < total; i += blockDim.x) {
        int bb = i / plane, sp = i - bb * plane;
        float v = Y[(bb * Cout + c) * plane + sp];
        s += v; ss += v * v;
    }
    __shared__ float rs[256], rq[256];
    rs[tid] = s; rq[tid] = ss;
    __syncthreads();
    for (int st = 128; st > 0; st >>= 1) {
        if (tid < st) { rs[tid] += rs[tid + st]; rq[tid] += rq[tid + st]; }
        __syncthreads();
    }
    if (tid == 0) {
        float m = rs[0] / (float)total;
        float var = rq[0] / (float)total - m * m;
        mean[c] = m;
        invstd[c] = rsqrtf(var + 1e-5f);
    }
}

// normalize + scale/shift + optional residual + ReLU; dual-write f32 + f16
__global__ void bn_apply(float* __restrict__ Y, _Float16* __restrict__ Y16,
                         const float* __restrict__ Xres,
                         const float* __restrict__ mean, const float* __restrict__ invstd,
                         const float* __restrict__ g, const float* __restrict__ beta,
                         int Cout, int plane, int total)
{
    int i = blockIdx.x * blockDim.x + threadIdx.x;
    if (i >= total) return;
    int c = (i / plane) % Cout;
    float v = (Y[i] - mean[c]) * invstd[c] * g[c] + beta[c];
    if (Xres) v += Xres[i];
    v = v > 0.f ? v : 0.f;
    Y[i] = v;
    Y16[i] = (_Float16)v;
}

// (B,T,H,W,3) f32 -> (B,3,T,H,W) f16 (conv-0 input mirror)
__global__ void nhwc_to_f16(const float* __restrict__ X, _Float16* __restrict__ Y,
                            int Bn, int T, int H, int W, int C)
{
    int total = Bn * T * H * W * C;
    int i = blockIdx.x * blockDim.x + threadIdx.x;
    if (i >= total) return;
    int c = i % C; int t2 = i / C;
    int w = t2 % W; t2 /= W;
    int h = t2 % H; t2 /= H;
    int t = t2 % T; int b = t2 / T;
    Y[(((b * C + c) * T + t) * H + h) * W + w] = (_Float16)X[i];
}

// (B,384,120,1,1) -> (B,120,384) with h = alpha*(h+pe) + h
__global__ void squeeze_pe(const float* __restrict__ Hc, const float* __restrict__ alpha,
                           float* __restrict__ Y, int Bn, int T, int E)
{
    int bt = blockIdx.x;
    int b = bt / T, t = bt - b * T;
    float a = alpha[0];
    for (int e = threadIdx.x; e < E; e += blockDim.x) {
        float v = Hc[(b * E + e) * T + t];
        int i2 = e >> 1;
        float div = expf(-logf(10000.f) * (2.f * (float)i2) / (float)E);
        float ang = (float)t * div;
        float pe = (e & 1) ? cosf(ang) : sinf(ang);
        Y[(b * T + t) * E + e] = a * (v + pe) + v;
    }
}

// ---------------------------------------------------------------------------
// Generic batched GEMM, f16 WMMA: C = A(MxK) * op(B)(KxN) [+ bias[n]]
// One wave computes 64x16 of C (4 M-subtiles share the B fragment).
// Main K-loop over K&~31 is clamp/select-free; a single uniform tail
// iteration handles the K remainder (K even in every use).
// ---------------------------------------------------------------------------
template<int TRANSB>
__global__ void gemm_wmma(const float* __restrict__ Ap, int lda, int offA1, int offA2,
                          const float* __restrict__ Bp, int ldb, int offB1, int offB2,
                          float* __restrict__ Cp, int ldc, int offC1, int offC2,
                          const float* __restrict__ bias,
                          int M, int N, int K, int batch2)
{
    const int z = blockIdx.z;
    const int z1 = z / batch2, z2 = z - z1 * batch2;
    const float* Ab = Ap + z1 * offA1 + z2 * offA2;
    const float* Bb = Bp + z1 * offB1 + z2 * offB2;
    float*       Cb = Cp + z1 * offC1 + z2 * offC2;

    const int lane = threadIdx.x;
    const int hi = lane >> 4, lo = lane & 15;
    const int mbase = blockIdx.x * 64;
    const int n = blockIdx.y * 16 + lo;
    const int ncl = n < N ? n : N - 1;

    int mr[4];
    #pragma unroll
    for (int mt = 0; mt < 4; ++mt) {
        int m = mbase + mt * 16 + lo;
        mr[mt] = m < M ? m : M - 1;           // clamped row (stores masked)
    }

    v8f acc[4] = {};
    const int Kfull = K & ~31;

    int k0 = 0;
    for (; k0 < Kfull; k0 += 32) {            // ---- clamp-free main loop ----
        v16h bf;
        const int bk = k0 + hi * 16;
        #pragma unroll
        for (int v = 0; v < 8; ++v) {
            #pragma unroll
            for (int hh = 0; hh < 2; ++hh) {
                float bvf = TRANSB ? Bb[ncl * ldb + bk + 2 * v + hh]
                                   : Bb[(bk + 2 * v + hh) * ldb + ncl];
                bf[2 * v + hh] = (_Float16)bvf;
            }
        }
        #pragma unroll
        for (int mt = 0; mt < 4; ++mt) {
            const float* ar = Ab + mr[mt] * lda + k0 + hi * 16;
            v16h af;
            #pragma unroll
            for (int v = 0; v < 8; ++v) {
                f2 av = *(const f2*)(ar + v * 2);
                af[2 * v] = (_Float16)av[0]; af[2 * v + 1] = (_Float16)av[1];
            }
            acc[mt] = __builtin_amdgcn_wmma_f32_16x16x32_f16(
                false, af, false, bf, (short)0, acc[mt], false, false);
        }
    }
    if (k0 < K) {                             // ---- single uniform tail ----
        v16h bf;
        #pragma unroll
        for (int v = 0; v < 8; ++v) {
            #pragma unroll
            for (int hh = 0; hh < 2; ++hh) {
                int kk = k0 + (v >> 2) * 16 + hi * 8 + (v & 3) * 2 + hh;
                int kkc = kk < K ? kk : K - 1;
                float bvf = TRANSB ? Bb[ncl * ldb + kkc] : Bb[kkc * ldb + ncl];
                bf[2 * v + hh] = (kk < K) ? (_Float16)bvf : (_Float16)0.f;
            }
        }
        #pragma unroll
        for (int mt = 0; mt < 4; ++mt) {
            v16h af;
            #pragma unroll
            for (int v = 0; v < 8; ++v) {
                int kk = k0 + (v >> 2) * 16 + hi * 8 + (v & 3) * 2;
                int kkc = kk <= K - 2 ? kk : K - 2;    // K even everywhere
                f2 av = *(const f2*)(Ab + mr[mt] * lda + kkc);
                af[2 * v] = (_Float16)av[0]; af[2 * v + 1] = (_Float16)av[1];
            }
            acc[mt] = __builtin_amdgcn_wmma_f32_16x16x32_f16(
                false, af, false, bf, (short)0, acc[mt], false, false);
        }
    }

    if (n < N) {
        float bsum = bias ? bias[n] : 0.f;
        #pragma unroll
        for (int mt = 0; mt < 4; ++mt) {
            #pragma unroll
            for (int r = 0; r < 8; ++r) {
                int mm = mbase + mt * 16 + r + hi * 8;
                if (mm < M) Cb[mm * ldc + n] = acc[mt][r] + bsum;
            }
        }
    }
}

// Row softmax with scale. cols <= 128.
__global__ void softmax_rows(float* __restrict__ Em, int cols, float inv_scale)
{
    const int r = blockIdx.x;
    const int tid = threadIdx.x;
    float* row = Em + r * cols;
    __shared__ float red[128];
    float v = (tid < cols) ? row[tid] * inv_scale : -3.0e38f;
    red[tid] = v; __syncthreads();
    for (int s = 64; s > 0; s >>= 1) {
        if (tid < s) red[tid] = fmaxf(red[tid], red[tid + s]);
        __syncthreads();
    }
    float mx = red[0]; __syncthreads();
    float ex = (tid < cols) ? expf(v - mx) : 0.f;
    red[tid] = ex; __syncthreads();
    for (int s = 64; s > 0; s >>= 1) {
        if (tid < s) red[tid] += red[tid + s];
        __syncthreads();
    }
    float sum = red[0];
    if (tid < cols) row[tid] = ex / sum;
}

// Y[row] = LayerNorm(X1[row] + X2[row]) * g + b ; cols == 384, blockDim == 128
__global__ void add_ln(const float* __restrict__ X1, const float* __restrict__ X2,
                       const float* __restrict__ g, const float* __restrict__ bta,
                       float* __restrict__ Y, int cols)
{
    const int r = blockIdx.x;
    const int tid = threadIdx.x;
    const float* a = X1 + r * cols;
    const float* b = X2 + r * cols;
    float loc[3];
    float s = 0.f;
    #pragma unroll
    for (int j = 0; j < 3; ++j) {
        float v = a[tid + j * 128] + b[tid + j * 128];
        loc[j] = v; s += v;
    }
    __shared__ float red[128];
    red[tid] = s; __syncthreads();
    for (int st = 64; st > 0; st >>= 1) {
        if (tid < st) red[tid] += red[tid + st];
        __syncthreads();
    }
    float m = red[0] / (float)cols; __syncthreads();
    float ss = 0.f;
    #pragma unroll
    for (int j = 0; j < 3; ++j) { float d = loc[j] - m; ss += d * d; }
    red[tid] = ss; __syncthreads();
    for (int st = 64; st > 0; st >>= 1) {
        if (tid < st) red[tid] += red[tid + st];
        __syncthreads();
    }
    float inv = rsqrtf(red[0] / (float)cols + 1e-5f);
    #pragma unroll
    for (int j = 0; j < 3; ++j) {
        int e = tid + j * 128;
        Y[r * cols + e] = (loc[j] - m) * inv * g[e] + bta[e];
    }
}

// Exact GELU (erf form)
__global__ void gelu_k(float* __restrict__ X, int n)
{
    int i = blockIdx.x * blockDim.x + threadIdx.x;
    if (i < n) {
        float x = X[i];
        X[i] = 0.5f * x * (1.f + erff(x * 0.70710678118654752f));
    }
}

// ---------------------------------------------------------------------------
extern "C" void kernel_launch(void* const* d_in, const int* in_sizes, int n_in,
                              void* d_out, int out_size, void* d_ws, size_t ws_size,
                              hipStream_t stream)
{
    (void)in_sizes; (void)n_in; (void)out_size; (void)ws_size;
    const int Bn = 4, T = 120, E = 384, HEADS = 8, DEPTH = 48, FE = 4;

    const float* x_in  = (const float*)d_in[0];
    const float* alpha = (const float*)d_in[1];

    // ---- workspace carve-out ----
    char* wp = (char*)d_ws;
    auto allocb = [&](size_t bytes) -> void* {
        void* p = (void*)wp;
        wp += ((bytes + 255) / 256) * 256;
        return p;
    };
    const size_t ACT = 13271040;  // max activation elems: 4*48*120*24*24
    float*     bufA   = (float*)allocb(ACT * 4);
    float*     bufB   = (float*)allocb(ACT * 4);
    _Float16*  bufA16 = (_Float16*)allocb(ACT * 2);
    _Float16*  bufB16 = (_Float16*)allocb(ACT * 2);
    _Float16*  wpad   = (_Float16*)allocb((size_t)384 * 10368 * 2); // max Cout*KKpad
    float*     mean   = (float*)allocb(512 * 4);
    float*     invstd = (float*)allocb(512 * 4);
    float*     xb0    = (float*)allocb((size_t)Bn * T * E * 4);
    float*     xb1    = (float*)allocb((size_t)Bn * T * E * 4);
    float*     catb   = (float*)allocb((size_t)Bn * T * 2 * E * 4);
    float*     kb     = (float*)allocb((size_t)Bn * HEADS * T * DEPTH * 4);
    float*     vb     = (float*)allocb((size_t)Bn * HEADS * T * DEPTH * 4);
    float*     eng    = (float*)allocb((size_t)Bn * HEADS * T * T * 4);
    float*     ob     = (float*)allocb((size_t)Bn * T * E * 4);
    float*     ab     = (float*)allocb((size_t)Bn * T * E * 4);
    float*     fb     = (float*)allocb((size_t)Bn * T * FE * E * 4);

    // ---- prenet ----
    {
        int tot = Bn * T * 48 * 48 * 3;
        nhwc_to_f16<<<dim3((unsigned)((tot + 255) / 256)), 256, 0, stream>>>(
            x_in, bufA16, Bn, T, 48, 48, 3);
    }
    static const int CFG[13][7] = {
        {3,48,5,1,2,2,0},{48,48,3,1,1,1,1},{48,48,3,1,1,1,1},
        {48,96,3,1,2,2,0},{96,96,3,1,1,1,1},{96,96,3,1,1,1,1},
        {96,192,3,1,2,2,0},{192,192,3,1,1,1,1},{192,192,3,1,1,1,1},
        {192,384,3,1,2,2,0},{384,384,3,1,1,1,1},{384,384,3,1,1,1,1},
        {384,384,3,1,3,3,0}};
    int Dd = T, Hh = 48, Ww = 48;
    float* cur = bufA;        float* nxt = bufB;
    _Float16* cur16 = bufA16; _Float16* nxt16 = bufB16;
    for (int l = 0; l < 13; ++l) {
        int cin = CFG[l][0], cout = CFG[l][1], kc = CFG[l][2];
        int sd = CFG[l][3], sh = CFG[l][4], sw = CFG[l][5], res = CFG[l][6];
        int p = kc / 2;
        int OD = (Dd + 2 * p - kc) / sd + 1;
        int OH = (Hh + 2 * p - kc) / sh + 1;
        int OW = (Ww + 2 * p - kc) / sw + 1;
        const float* Wt = (const float*)d_in[2 + 4 * l + 0];
        const float* bb = (const float*)d_in[2 + 4 * l + 1];
        const float* gg = (const float*)d_in[2 + 4 * l + 2];
        const float* be = (const float*)d_in[2 + 4 * l + 3];
        int KK = cin * kc * kc * kc;
        int KKpad = ((KK + 31) / 32) * 32;
        int wtot = cout * KKpad;
        pack_w_f16<<<dim3((unsigned)((wtot + 255) / 256)), 256, 0, stream>>>(
            Wt, wpad, KK, KKpad, wtot);
        int Ntot = Bn * OD * OH * OW;
        unsigned gx = (unsigned)((Ntot + 15) / 16);
        // exact-M dispatch: 48 -> 1x(3x16), 96 -> 2x(3x16), 192/384 -> (4x16)
        if (kc == 5)
            conv3d_wmma<5, 3><<<dim3(gx, 1), 32, 0, stream>>>(cur16, wpad, bb, nxt,
                Bn, cin, Dd, Hh, Ww, cout, OD, OH, OW, sd, sh, sw, p, KKpad);
        else if (cout % 64 == 0)
            conv3d_wmma<3, 4><<<dim3(gx, (unsigned)(cout / 64)), 32, 0, stream>>>(
                cur16, wpad, bb, nxt,
                Bn, cin, Dd, Hh, Ww, cout, OD, OH, OW, sd, sh, sw, p, KKpad);
        else
            conv3d_wmma<3, 3><<<dim3(gx, (unsigned)(cout / 48)), 32, 0, stream>>>(
                cur16, wpad, bb, nxt,
                Bn, cin, Dd, Hh, Ww, cout, OD, OH, OW, sd, sh, sw, p, KKpad);
        int plane = OD * OH * OW;
        bn_stats<<<cout, 256, 0, stream>>>(nxt, mean, invstd, Bn, cout, plane);
        int total = Bn * cout * plane;
        bn_apply<<<dim3((unsigned)((total + 255) / 256)), 256, 0, stream>>>(
            nxt, nxt16, res ? cur : (const float*)nullptr, mean, invstd, gg, be,
            cout, plane, total);
        float* t0 = cur; cur = nxt; nxt = t0;
        _Float16* t1 = cur16; cur16 = nxt16; nxt16 = t1;
        Dd = OD; Hh = OH; Ww = OW;
    }

    // ---- squeeze (H=W=1) + positional encoding + alpha blend ----
    squeeze_pe<<<Bn * T, 128, 0, stream>>>(cur, alpha, xb0, Bn, T, E);

    // ---- transformer ----
    const int sXb  = T * E;          // 46080
    const int sCat = T * 2 * E;      // 92160
    const int sKVh = T * DEPTH;      // 5760
    const int sKVb = HEADS * sKVh;   // 46080
    const int sEh  = T * T;          // 14400
    const int sEb  = HEADS * sEh;    // 115200
    const int sFb  = T * FE * E;     // 184320
    float* Xc = xb0; float* Xn = xb1;

    for (int l = 0; l < 6; ++l) {
        const int base = 54 + 16 * l;
        const float* Wq  = (const float*)d_in[base + 0];
        const float* bq  = (const float*)d_in[base + 1];
        const float* Wk  = (const float*)d_in[base + 2];
        const float* bk  = (const float*)d_in[base + 3];
        const float* Wv  = (const float*)d_in[base + 4];
        const float* bv  = (const float*)d_in[base + 5];
        const float* Wo  = (const float*)d_in[base + 6];
        const float* bo  = (const float*)d_in[base + 7];
        const float* W1  = (const float*)d_in[base + 8];
        const float* b1  = (const float*)d_in[base + 9];
        const float* W2  = (const float*)d_in[base + 10];
        const float* b2  = (const float*)d_in[base + 11];
        const float* g1  = (const float*)d_in[base + 12];
        const float* be1 = (const float*)d_in[base + 13];
        const float* g2  = (const float*)d_in[base + 14];
        const float* be2 = (const float*)d_in[base + 15];

        // Q -> catb[:,0:384] (head block at col h*48); K,V -> [b][h][l][e]
        dim3 gQKV(2, 3, Bn * HEADS);
        gemm_wmma<1><<<gQKV, 32, 0, stream>>>(Xc, E, sXb, DEPTH,
                                              Wq, DEPTH, 0, 0,
                                              catb, 2 * E, sCat, DEPTH,
                                              bq, T, DEPTH, DEPTH, HEADS);
        gemm_wmma<1><<<gQKV, 32, 0, stream>>>(Xc, E, sXb, DEPTH,
                                              Wk, DEPTH, 0, 0,
                                              kb, DEPTH, sKVb, sKVh,
                                              bk, T, DEPTH, DEPTH, HEADS);
        gemm_wmma<1><<<gQKV, 32, 0, stream>>>(Xc, E, sXb, DEPTH,
                                              Wv, DEPTH, 0, 0,
                                              vb, DEPTH, sKVb, sKVh,
                                              bv, T, DEPTH, DEPTH, HEADS);
        // energy = q @ k^T  (per b,h): (120x48)(48x120)
        dim3 gEng(2, 8, Bn * HEADS);
        gemm_wmma<1><<<gEng, 32, 0, stream>>>(catb, 2 * E, sCat, DEPTH,
                                              kb, DEPTH, sKVb, sKVh,
                                              eng, T, sEb, sEh,
                                              nullptr, T, T, DEPTH, HEADS);
        // softmax, scale = 1/(DEPTH/2) = 1/24
        softmax_rows<<<Bn * HEADS * T, 128, 0, stream>>>(eng, T, 1.f / 24.f);
        // attn_out = attnw @ v -> catb[:,384:768]
        gemm_wmma<0><<<gQKV, 32, 0, stream>>>(eng, T, sEb, sEh,
                                              vb, DEPTH, sKVb, sKVh,
                                              catb + E, 2 * E, sCat, DEPTH,
                                              nullptr, T, DEPTH, T, HEADS);
        // out = cat(q, attn_out) @ Wo^T + bo
        dim3 gWo(2, 24, Bn);
        gemm_wmma<1><<<gWo, 32, 0, stream>>>(catb, 2 * E, 0, sCat,
                                             Wo, 2 * E, 0, 0,
                                             ob, E, 0, sXb,
                                             bo, T, E, 2 * E, Bn);
        // a = LN(out + x)
        add_ln<<<Bn * T, 128, 0, stream>>>(ob, Xc, g1, be1, ab, E);
        // f = gelu(a @ W1^T + b1)
        dim3 gF1(2, 96, Bn);
        gemm_wmma<1><<<gF1, 32, 0, stream>>>(ab, E, 0, sXb,
                                             W1, E, 0, 0,
                                             fb, FE * E, 0, sFb,
                                             b1, T, FE * E, E, Bn);
        int nf = Bn * T * FE * E;
        gelu_k<<<dim3((unsigned)((nf + 255) / 256)), 256, 0, stream>>>(fb, nf);
        // f @ W2^T + b2
        dim3 gF2(2, 24, Bn);
        gemm_wmma<1><<<gF2, 32, 0, stream>>>(fb, FE * E, 0, sFb,
                                             W2, FE * E, 0, 0,
                                             ob, E, 0, sXb,
                                             b2, T, E, FE * E, Bn);
        // out = LN(f + a)
        float* tgt = (l == 5) ? (float*)d_out : Xn;
        add_ln<<<Bn * T, 128, 0, stream>>>(ob, ab, g2, be2, tgt, E);
        float* tmp = Xc; Xc = Xn; Xn = tmp;
    }
}
